// LookUpGCN_7224134992211
// MI455X (gfx1250) — matrix operand order repaired and checked by reading, not compiled, not run on hardware
//
#include <hip/hip_runtime.h>
#include <hip/hip_bf16.h>

typedef __attribute__((ext_vector_type(16))) _Float16 v16h;
typedef __attribute__((ext_vector_type(8)))  float    v8f;

#define D 128
#define NEG_SLOPE 0.2f
#define LN_EPS 1e-5f

// ---------------------------------------------------------------------------
// Utility: full-wave (32-lane) sum reduction
__device__ __forceinline__ float wave_sum(float v) {
#pragma unroll
    for (int off = 16; off >= 1; off >>= 1) v += __shfl_xor(v, off);
    return v;
}

// ---------------------------------------------------------------------------
// Mean of edge weights: per-wave partial + atomicAdd into wsum[0]
__global__ void edge_mean_kernel(const float* __restrict__ ew, int nE, float* wsum) {
    int i = blockIdx.x * 256 + threadIdx.x;
    float v = (i < nE) ? ew[i] : 0.0f;
    v = wave_sum(v);
    if ((threadIdx.x & 31) == 0) atomicAdd(wsum, v);
}

// ---------------------------------------------------------------------------
// x[i] = emb[node_ids[i]]  (one float4 per thread)
__global__ void gather_kernel(const int* __restrict__ ids, const float* __restrict__ emb,
                              float* __restrict__ x, int nN) {
    int t = blockIdx.x * 256 + threadIdx.x;
    int total = nN * (D / 4);
    if (t >= total) return;
    int i = t >> 5;          // node
    int c = t & 31;          // float4 slot within row
    ((float4*)x)[(size_t)i * 32 + c] =
        ((const float4*)emb)[(size_t)ids[i] * 32 + c];
}

// ---------------------------------------------------------------------------
// Pack W (128x128 f32, row-major K x N) into f16 WMMA B-fragment order:
//   frag[(ct*4 + kc)*32 + lane][j] = f16( W[(kc*32 + (lane>>4)*16 + j)*D + ct*16 + (lane&15)] )
// One thread per fragment-lane (8 colTiles * 4 kchunks * 32 lanes = 1024).
__global__ void pack_w_kernel(const float* __restrict__ Wl, const float* __restrict__ Wr,
                              v16h* __restrict__ wbl, v16h* __restrict__ wbr) {
    int t = blockIdx.x * 256 + threadIdx.x;
    if (t >= 8 * 4 * 32) return;
    const float* W = blockIdx.y ? Wr : Wl;
    v16h*        o = blockIdx.y ? wbr : wbl;
    int lane = t & 31, kc = (t >> 5) & 3, ct = t >> 7;
    int hl = lane & 15, half = lane >> 4;
    const float* p = W + (size_t)(kc * 32 + half * 16) * D + ct * 16 + hl;
    v16h b;
#pragma unroll
    for (int j = 0; j < 16; ++j) b[j] = (_Float16)p[j * D];
    o[t] = b;
}

// ---------------------------------------------------------------------------
// Pack x (nN x 128 f32, row-major M x K) into f16 WMMA A-fragment order:
// lane covers K = kc*32 + half*16 + {0..15} of row rowTile+(lane&15).
// One thread per fragment-lane: total = tiles*4*32.
__global__ void pack_x_kernel(const float* __restrict__ x, v16h* __restrict__ xa,
                              int nN, int tiles) {
    int t = blockIdx.x * 256 + threadIdx.x;
    if (t >= tiles * 4 * 32) return;
    int lane = t & 31, kc = (t >> 5) & 3, tile = t >> 7;
    int hl = lane & 15, half = lane >> 4;
    int row = tile * 16 + hl;
    if (row >= nN) row = nN - 1;          // clamp (stores guarded in GEMM)
    const float* p = x + (size_t)row * D + kc * 32 + half * 16;
    v16h a;
#pragma unroll
    for (int j = 0; j < 8; ++j) {
        a[j]     = (_Float16)p[j];
        a[8 + j] = (_Float16)p[8 + j];
    }
    xa[t] = a;
}

// ---------------------------------------------------------------------------
// Dual GEMM with WMMA from pre-packed fragments: out = x @ W + b.
// grid.x = row tiles, grid.y = 2 (left/right). 8 waves/block; wave w owns the
// 16x16 tile at column 16*w. K=128 -> 4 WMMAs; each fed by two 32B frag loads.
__global__ __launch_bounds__(256)
void gemm_bias_wmma_kernel(const v16h* __restrict__ xa,
                           const v16h* __restrict__ wbl, const float* __restrict__ bl,
                           const v16h* __restrict__ wbr, const float* __restrict__ br,
                           float* __restrict__ xl, float* __restrict__ xr, int nN) {
    const v16h*  wb   = blockIdx.y ? wbr : wbl;
    const float* bias = blockIdx.y ? br : bl;
    float*       out  = blockIdx.y ? xr : xl;

    const int tile = blockIdx.x;
    const int ct   = threadIdx.x >> 5;     // colTile 0..7
    const int lane = threadIdx.x & 31;
    const int half = lane >> 4;
    const int hl   = lane & 15;

    v8f c = {};
#pragma unroll
    for (int kc = 0; kc < 4; ++kc) {
        v16h a = xa[((size_t)tile * 4 + kc) * 32 + lane];
        v16h b = wb[((size_t)ct * 4 + kc) * 32 + lane];
        c = __builtin_amdgcn_wmma_f32_16x16x32_f16(false, a, false, b,
                                                   (short)0, c, false, false);
    }

    const float bv = bias[ct * 16 + hl];
    // C layout: VGPR r -> M = tile*16 + r (lanes 0-15) / + r+8 (lanes 16-31)
    float* obase = out + (size_t)(tile * 16 + half * 8) * D + ct * 16 + hl;
    if (tile * 16 + 16 <= nN) {             // full tile: branchless store clause
#pragma unroll
        for (int r = 0; r < 8; ++r) obase[(size_t)r * D] = c[r] + bv;
    } else {                                // ragged tail tile (unused for nN%16==0)
#pragma unroll
        for (int r = 0; r < 8; ++r) {
            int m = tile * 16 + r + half * 8;
            if (m < nN) out[(size_t)m * D + ct * 16 + hl] = c[r] + bv;
        }
    }
}

// ---------------------------------------------------------------------------
// Pass 1: per-edge attention logit + segment max (uint-key atomicMax).
// One wave per edge; lane holds one float4 of the 128-d row.
__global__ __launch_bounds__(256)
void edge_logit_kernel(const int* __restrict__ ei, const float* __restrict__ ew,
                       const float* __restrict__ wsum,
                       const float* __restrict__ xl, const float* __restrict__ xr,
                       const float* __restrict__ We, const float* __restrict__ att,
                       float* __restrict__ logits, unsigned* __restrict__ segmax,
                       int nE, int nN) {
    int e = blockIdx.x * 8 + (threadIdx.x >> 5);
    if (e >= nE + nN) return;
    int lane = threadIdx.x & 31;

    int s, d; float w;
    if (e < nE) { s = ei[e]; d = ei[nE + e]; w = ew[e]; }
    else        { s = d = e - nE;            w = wsum[0] / (float)nE; }

    float4 av = ((const float4*)(xl + (size_t)s * D))[lane];
    float4 bv = ((const float4*)(xr + (size_t)d * D))[lane];
    float4 wv = ((const float4*)We)[lane];
    float4 tv = ((const float4*)att)[lane];

    float m0 = av.x + bv.x + w * wv.x; m0 = m0 > 0.0f ? m0 : NEG_SLOPE * m0;
    float m1 = av.y + bv.y + w * wv.y; m1 = m1 > 0.0f ? m1 : NEG_SLOPE * m1;
    float m2 = av.z + bv.z + w * wv.z; m2 = m2 > 0.0f ? m2 : NEG_SLOPE * m2;
    float m3 = av.w + bv.w + w * wv.w; m3 = m3 > 0.0f ? m3 : NEG_SLOPE * m3;
    float acc = m0 * tv.x + m1 * tv.y + m2 * tv.z + m3 * tv.w;
    acc = wave_sum(acc);

    if (lane == 0) {
        logits[e] = acc;
        unsigned k = __float_as_uint(acc);
        k = (k & 0x80000000u) ? ~k : (k | 0x80000000u);   // monotone float key
        atomicMax(segmax + d, k);
    }
}

// ---------------------------------------------------------------------------
// Pass 2: e = exp(logit - max[dst]); segsum[dst] += e  (one thread per edge)
__global__ void edge_exp_kernel(const int* __restrict__ ei,
                                float* __restrict__ logits,
                                const unsigned* __restrict__ segmax,
                                float* __restrict__ segsum, int nE, int nN) {
    int e = blockIdx.x * 256 + threadIdx.x;
    if (e >= nE + nN) return;
    int d = (e < nE) ? ei[nE + e] : (e - nE);
    unsigned k = segmax[d];
    float mx = (k & 0x80000000u) ? __uint_as_float(k & 0x7FFFFFFFu)
                                 : __uint_as_float(~k);
    float ev = __expf(logits[e] - mx);
    logits[e] = ev;
    atomicAdd(segsum + d, ev);
}

// ---------------------------------------------------------------------------
// Pass 3: accum[dst] += xl[src] * (e / segsum[dst])  (wave per edge, f32 atomics)
__global__ __launch_bounds__(256)
void edge_scatter_kernel(const int* __restrict__ ei, const float* __restrict__ expv,
                         const float* __restrict__ segsum, const float* __restrict__ xl,
                         float* __restrict__ accum, int nE, int nN) {
    int e = blockIdx.x * 8 + (threadIdx.x >> 5);
    if (e >= nE + nN) return;
    int lane = threadIdx.x & 31;
    int s, d;
    if (e < nE) { s = ei[e]; d = ei[nE + e]; }
    else        { s = d = e - nE; }
    float coef = expv[e] / segsum[d];
    float4 v = ((const float4*)(xl + (size_t)s * D))[lane];
    float* o = accum + (size_t)d * D + lane * 4;
    atomicAdd(o + 0, v.x * coef);
    atomicAdd(o + 1, v.y * coef);
    atomicAdd(o + 2, v.z * coef);
    atomicAdd(o + 3, v.w * coef);
}

// ---------------------------------------------------------------------------
// Fused residual + LayerNorm: xout = LN(xin + accum + bias) * g + be
// One wave per node.
__global__ __launch_bounds__(256)
void residual_ln_kernel(const float* __restrict__ xin, const float* __restrict__ accum,
                        const float* __restrict__ bias, const float* __restrict__ g,
                        const float* __restrict__ be, float* __restrict__ xout, int nN) {
    int i = blockIdx.x * 8 + (threadIdx.x >> 5);
    if (i >= nN) return;
    int lane = threadIdx.x & 31;

    float4 xv = ((const float4*)(xin   + (size_t)i * D))[lane];
    float4 av = ((const float4*)(accum + (size_t)i * D))[lane];
    float4 bv = ((const float4*)bias)[lane];
    float4 y  = make_float4(xv.x + av.x + bv.x, xv.y + av.y + bv.y,
                            xv.z + av.z + bv.z, xv.w + av.w + bv.w);

    float s  = y.x + y.y + y.z + y.w;
    float ss = y.x * y.x + y.y * y.y + y.z * y.z + y.w * y.w;
    s  = wave_sum(s);
    ss = wave_sum(ss);
    float mu  = s * (1.0f / D);
    float var = ss * (1.0f / D) - mu * mu;
    float r   = rsqrtf(var + LN_EPS);

    float4 gg = ((const float4*)g)[lane];
    float4 bb = ((const float4*)be)[lane];
    float4 o  = make_float4((y.x - mu) * r * gg.x + bb.x,
                            (y.y - mu) * r * gg.y + bb.y,
                            (y.z - mu) * r * gg.z + bb.z,
                            (y.w - mu) * r * gg.w + bb.w);
    ((float4*)(xout + (size_t)i * D))[lane] = o;
}

// ---------------------------------------------------------------------------
extern "C" void kernel_launch(void* const* d_in, const int* in_sizes, int n_in,
                              void* d_out, int out_size, void* d_ws, size_t ws_size,
                              hipStream_t stream) {
    const int*   node_ids = (const int*)d_in[0];
    const int*   ei       = (const int*)d_in[1];
    const float* ew       = (const float*)d_in[2];
    const float* emb      = (const float*)d_in[3];

    const int nN    = in_sizes[0];
    const int nE    = in_sizes[2];
    const int Etot  = nE + nN;
    const int tiles = (nN + 15) / 16;

    // workspace layout: x | xl | xr | xa(A-frags f16) | wbl | wbr | segmax | segsum | wsum | logits
    float*    x      = (float*)d_ws;
    float*    xl     = x  + (size_t)nN * D;
    float*    xr     = xl + (size_t)nN * D;
    v16h*     xa     = (v16h*)(xr + (size_t)nN * D);            // tiles*4*32 frags
    v16h*     wbl    = xa  + (size_t)tiles * 4 * 32;
    v16h*     wbr    = wbl + 8 * 4 * 32;
    unsigned* segmax = (unsigned*)(wbr + 8 * 4 * 32);
    float*    segsum = (float*)(segmax + nN);
    float*    wsum   = segsum + nN;
    float*    logits = wsum + 1;
    float*    accum  = (float*)d_out;   // doubles as segment-sum accumulator

    // zero segmax | segsum | wsum (contiguous) and accum
    hipMemsetAsync(segmax, 0, ((size_t)2 * nN + 1) * sizeof(float), stream);
    hipMemsetAsync(accum,  0, (size_t)nN * D * sizeof(float), stream);

    edge_mean_kernel<<<(nE + 255) / 256, 256, 0, stream>>>(ew, nE, wsum);
    gather_kernel<<<(nN * (D / 4) + 255) / 256, 256, 0, stream>>>(node_ids, emb, x, nN);

    for (int L = 0; L < 2; ++L) {
        const float* Wl   = (const float*)d_in[4 + L * 9 + 0];
        const float* bl   = (const float*)d_in[4 + L * 9 + 1];
        const float* Wr   = (const float*)d_in[4 + L * 9 + 2];
        const float* br   = (const float*)d_in[4 + L * 9 + 3];
        const float* We   = (const float*)d_in[4 + L * 9 + 4];
        const float* att  = (const float*)d_in[4 + L * 9 + 5];
        const float* bias = (const float*)d_in[4 + L * 9 + 6];
        const float* g    = (const float*)d_in[4 + L * 9 + 7];
        const float* be   = (const float*)d_in[4 + L * 9 + 8];

        if (L) {  // reset segment state + accumulator for layer 2 (keep wsum)
            hipMemsetAsync(segmax, 0, (size_t)2 * nN * sizeof(float), stream);
            hipMemsetAsync(accum,  0, (size_t)nN * D * sizeof(float), stream);
        }

        // pack operands into WMMA fragment order (f16)
        pack_w_kernel<<<dim3(4, 2), 256, 0, stream>>>(Wl, Wr, wbl, wbr);
        pack_x_kernel<<<(tiles * 128 + 255) / 256, 256, 0, stream>>>(x, xa, nN, tiles);

        dim3 ggrid(tiles, 2);
        gemm_bias_wmma_kernel<<<ggrid, 256, 0, stream>>>(xa, wbl, bl, wbr, br, xl, xr, nN);

        edge_logit_kernel<<<(Etot + 7) / 8, 256, 0, stream>>>(
            ei, ew, wsum, xl, xr, We, att, logits, segmax, nE, nN);
        edge_exp_kernel<<<(Etot + 255) / 256, 256, 0, stream>>>(
            ei, logits, segmax, segsum, nE, nN);
        edge_scatter_kernel<<<(Etot + 7) / 8, 256, 0, stream>>>(
            ei, logits, segsum, xl, accum, nE, nN);

        residual_ln_kernel<<<(nN + 7) / 8, 256, 0, stream>>>(
            x, accum, bias, g, be, (L == 0) ? x : (float*)d_out, nN);
    }
}
